// EncoderLayer_31722628448806
// MI455X (gfx1250) — matrix operand order repaired and checked by reading, not compiled, not run on hardware
//
#include <hip/hip_runtime.h>
#include <cstdint>
#include <cstddef>

// ---------------- problem constants ----------------
#define BATCH     4
#define SEQ       2048
#define DMODEL    1024
#define DSTATE    64
#define DCONV     4
#define HEADDIM   64
#define DINNER    2048               // EXPAND * DMODEL
#define NHEADS    32                 // DINNER / HEADDIM
#define CONVDIM   2176               // DINNER + 2*DSTATE
#define DINPROJ   4256               // 2*DINNER + 2*DSTATE + NHEADS
#define NPAD      4352               // DINPROJ padded up to multiple of 128
#define MTOT      (BATCH*SEQ)        // 8192
#define EPSV      1e-5f

// chunked-scan constants
#define QLEN      64                 // chunk length
#define CHUNKS    (SEQ/QLEN)         // 32
#define NCHUNK    (BATCH*NHEADS*CHUNKS)  // 4096
#define LDS2      72                 // LDS row stride for 64-wide bf16 tiles (144B, 16B aligned)

typedef __attribute__((ext_vector_type(16))) __bf16  v16bf;
typedef __attribute__((ext_vector_type(8)))  float   v8f;
typedef __attribute__((ext_vector_type(4)))  unsigned int u32x4;

union FragAB { u32x4 u[2]; v16bf v; };

__device__ __forceinline__ uint16_t f32_to_bf16(float f) {
  uint32_t u = __builtin_bit_cast(uint32_t, f);
  uint32_t r = u + 0x7FFFu + ((u >> 16) & 1u);   // round-to-nearest-even
  return (uint16_t)(r >> 16);
}

__device__ __forceinline__ float sigmoidf_(float v) { return 1.f / (1.f + __expf(-v)); }

// ---------------- fp32 -> bf16 conversion ----------------
__global__ void cvt_bf16_kernel(const float* __restrict__ src,
                                uint16_t* __restrict__ dst, int n) {
  int i = blockIdx.x * blockDim.x + threadIdx.x;
  if (i < n) dst[i] = f32_to_bf16(src[i]);
}

__global__ void cvt_bf16_pad_kernel(const float* __restrict__ src,
                                    uint16_t* __restrict__ dst,
                                    int n_valid, int n_total) {
  int i = blockIdx.x * blockDim.x + threadIdx.x;
  if (i < n_total) dst[i] = (i < n_valid) ? f32_to_bf16(src[i]) : (uint16_t)0;
}

// ---------------- WMMA bf16 GEMM: C[M,N] = A[M,K] * B[N,K]^T (+residual) ----------------
#define BM  128
#define BN  128
#define BK  32
#define LDT 40

__global__ __launch_bounds__(256)
void gemm_bf16_wmma_kernel(const uint16_t* __restrict__ A,
                           const uint16_t* __restrict__ B,
                           float* __restrict__ C,
                           const float* __restrict__ residual,
                           int N, int K) {
  __shared__ __attribute__((aligned(16))) uint16_t As[BM * LDT];
  __shared__ __attribute__((aligned(16))) uint16_t Bs[BN * LDT];

  const int t     = threadIdx.x;
  const int lane  = t & 31;
  const int wave  = t >> 5;
  const int wm    = wave & 1;
  const int wn    = wave >> 1;
  const int mBase = wm * 64;
  const int nBase = wn * 32;
  const int lm    = lane & 15;
  const int kg    = lane >> 4;

  const int blockM = blockIdx.x * BM;
  const int blockN = blockIdx.y * BN;

  v8f acc[4][2];
  const v8f vzero = {0.f,0.f,0.f,0.f,0.f,0.f,0.f,0.f};
#pragma unroll
  for (int i = 0; i < 4; ++i)
#pragma unroll
    for (int j = 0; j < 2; ++j) acc[i][j] = vzero;

  for (int k0 = 0; k0 < K; k0 += BK) {
#pragma unroll
    for (int pass = 0; pass < 2; ++pass) {
      int idx = pass * 256 + t;
      int row = idx >> 2;
      int cg  = (idx & 3) << 3;
      *(u32x4*)&As[row * LDT + cg] =
          *(const u32x4*)(A + (size_t)(blockM + row) * K + k0 + cg);
    }
#pragma unroll
    for (int pass = 0; pass < 2; ++pass) {
      int idx = pass * 256 + t;
      int row = idx >> 2;
      int cg  = (idx & 3) << 3;
      *(u32x4*)&Bs[row * LDT + cg] =
          *(const u32x4*)(B + (size_t)(blockN + row) * K + k0 + cg);
    }
    __syncthreads();

    FragAB af[4], bfg[2];
#pragma unroll
    for (int i = 0; i < 4; ++i) {
      const uint16_t* p = &As[(mBase + 16 * i + lm) * LDT + kg * 8];
      af[i].u[0] = *(const u32x4*)p;
      af[i].u[1] = *(const u32x4*)(p + 16);
    }
#pragma unroll
    for (int j = 0; j < 2; ++j) {
      const uint16_t* p = &Bs[(nBase + 16 * j + lm) * LDT + kg * 8];
      bfg[j].u[0] = *(const u32x4*)p;
      bfg[j].u[1] = *(const u32x4*)(p + 16);
    }

#pragma unroll
    for (int i = 0; i < 4; ++i)
#pragma unroll
      for (int j = 0; j < 2; ++j)
        acc[i][j] = __builtin_amdgcn_wmma_f32_16x16x32_bf16(
            false, af[i].v, false, bfg[j].v, (short)0, acc[i][j], false, false);

    __syncthreads();
  }

#pragma unroll
  for (int i = 0; i < 4; ++i)
#pragma unroll
    for (int j = 0; j < 2; ++j) {
      int n = blockN + nBase + 16 * j + lm;
#pragma unroll
      for (int v = 0; v < 8; ++v) {
        int m = blockM + mBase + 16 * i + v + 8 * kg;
        float val = acc[i][j][v];
        if (residual) val += residual[(size_t)m * N + n];
        C[(size_t)m * N + n] = val;
      }
    }
}

// ---------------- wave-level 64x64 (K=64) bf16 GEMM helper over LDS tiles ----------------
// OUT tiles for this wave: (mt, nt0) and (mt, nt0+1); A,B row-major [64][LDS2] bf16.
__device__ __forceinline__ void wave_gemm64(const uint16_t* As, const uint16_t* Bs,
                                            int mt, int nt0, v8f acc[2], int lane) {
  const int lm = lane & 15;
  const int kg = lane >> 4;
#pragma unroll
  for (int k0 = 0; k0 < 64; k0 += 32) {
    FragAB a;
    const uint16_t* pa = As + (mt * 16 + lm) * LDS2 + k0 + kg * 8;
    a.u[0] = *(const u32x4*)pa;
    a.u[1] = *(const u32x4*)(pa + 16);
#pragma unroll
    for (int j = 0; j < 2; ++j) {
      FragAB b;
      const uint16_t* pb = Bs + ((nt0 + j) * 16 + lm) * LDS2 + k0 + kg * 8;
      b.u[0] = *(const u32x4*)pb;
      b.u[1] = *(const u32x4*)(pb + 16);
      acc[j] = __builtin_amdgcn_wmma_f32_16x16x32_bf16(
          false, a.v, false, b.v, (short)0, acc[j], false, false);
    }
  }
}

// ---------------- depthwise causal conv (width 4) + SiLU ----------------
__global__ void conv_silu_kernel(const float* __restrict__ zx,
                                 const float* __restrict__ conv_w,
                                 const float* __restrict__ conv_b,
                                 float* __restrict__ xconv) {
  int i = blockIdx.x * blockDim.x + threadIdx.x;
  const int total = MTOT * CONVDIM;
  if (i >= total) return;
  int c = i % CONVDIM;
  int l = (i / CONVDIM) % SEQ;
  int b = i / (CONVDIM * SEQ);
  float acc = conv_b[c];
#pragma unroll
  for (int j = 0; j < DCONV; ++j) {
    int ls = l - (DCONV - 1) + j;
    if (ls >= 0)
      acc += conv_w[c * DCONV + j] *
             zx[(size_t)(b * SEQ + ls) * NPAD + DINNER + c];
  }
  xconv[i] = acc * sigmoidf_(acc);
}

// ---------------- dt = softplus(raw+bias); ldA = dt * (-exp(A_log)) (exact log of dA) ----
__global__ void dt_kernel(const float* __restrict__ zx,
                          const float* __restrict__ dt_bias,
                          const float* __restrict__ A_log,
                          float* __restrict__ dt_out,
                          float* __restrict__ ldA_out) {
  int i = blockIdx.x * blockDim.x + threadIdx.x;
  const int total = MTOT * NHEADS;
  if (i >= total) return;
  int h = i % NHEADS;
  int r = i / NHEADS;
  float raw = zx[(size_t)r * NPAD + (DINNER + CONVDIM) + h] + dt_bias[h];
  float sp  = (raw > 20.f) ? raw : log1pf(expf(raw));
  dt_out[i]  = sp;
  ldA_out[i] = sp * (-expf(A_log[h]));
}

// ---------------- chunked scan pass A: per-chunk cumsum + chunk state Hc ----------------
// block = (b*NHEADS+h)*CHUNKS + c.  Hc[p][n] = sum_s exp(cs63-cs[s])*dt[s]*x[s][p]*B[s][n]
__global__ __launch_bounds__(256)
void chunk_state_kernel(const float* __restrict__ xconv,
                        const float* __restrict__ dtv,
                        const float* __restrict__ ldAv,
                        float* __restrict__ cs_g,
                        float* __restrict__ Hc_g) {
  const int blk = blockIdx.x;
  const int c   = blk % CHUNKS;
  const int bh  = blk / CHUNKS;
  const int hh  = bh % NHEADS;
  const int b   = bh / NHEADS;
  const int t   = threadIdx.x;
  const int lane = t & 31, wave = t >> 5;

  __shared__ float csL[QLEN];
  __shared__ float dtL[QLEN];
  __shared__ __attribute__((aligned(16))) uint16_t xT[QLEN * LDS2];   // [p][s]
  __shared__ __attribute__((aligned(16))) uint16_t BTw[QLEN * LDS2];  // [n][s]

  const size_t row0 = (size_t)b * SEQ + c * QLEN;

  if (t < QLEN) {
    csL[t] = ldAv[(row0 + t) * NHEADS + hh];
    dtL[t] = dtv[(row0 + t) * NHEADS + hh];
  }
  __syncthreads();
  // inclusive cumsum (Hillis-Steele) over 64 entries
  for (int sdist = 1; sdist < QLEN; sdist <<= 1) {
    float v = 0.f;
    if (t < QLEN && t >= sdist) v = csL[t - sdist];
    __syncthreads();
    if (t < QLEN) csL[t] += v;
    __syncthreads();
  }
  if (t < QLEN) cs_g[(size_t)blk * QLEN + t] = csL[t];
  const float cs63 = csL[QLEN - 1];

  // stage transposed tiles (weights folded into B)
  for (int e = t; e < QLEN * QLEN; e += 256) {
    int s = e >> 6;
    int d = e & 63;
    float xv = xconv[(row0 + s) * CONVDIM + hh * HEADDIM + d];
    xT[d * LDS2 + s] = f32_to_bf16(xv);
    float bv = xconv[(row0 + s) * CONVDIM + DINNER + d];
    float w  = dtL[s] * expf(cs63 - csL[s]);     // <= 1, safe
    BTw[d * LDS2 + s] = f32_to_bf16(bv * w);
  }
  __syncthreads();

  const v8f vzero = {0.f,0.f,0.f,0.f,0.f,0.f,0.f,0.f};
  v8f acc[2] = {vzero, vzero};
  const int mt = wave & 3, nt0 = (wave >> 2) * 2;
  wave_gemm64(xT, BTw, mt, nt0, acc, lane);

  const int lm = lane & 15, kg = lane >> 4;
  float* out = Hc_g + (size_t)blk * (HEADDIM * DSTATE);
#pragma unroll
  for (int j = 0; j < 2; ++j) {
    int n = (nt0 + j) * 16 + lm;
#pragma unroll
    for (int v = 0; v < 8; ++v) {
      int p = mt * 16 + v + 8 * kg;
      out[p * DSTATE + n] = acc[j][v];
    }
  }
}

// ---------------- chunked scan pass B: 32-step inter-chunk recurrence (fp32) -------------
// block = b*NHEADS+h; state h0[p][n] kept in registers; stores chunk-start state (bf16).
__global__ __launch_bounds__(256)
void chunk_recur_kernel(const float* __restrict__ Hc_g,
                        const float* __restrict__ cs_g,
                        uint16_t* __restrict__ Hp_g) {
  const int bh = blockIdx.x;
  const int t  = threadIdx.x;
  float S[16];
#pragma unroll
  for (int i = 0; i < 16; ++i) S[i] = 0.f;

  for (int c = 0; c < CHUNKS; ++c) {
    size_t blk = (size_t)bh * CHUNKS + c;
    const float* hc = Hc_g + blk * (HEADDIM * DSTATE);
    uint16_t*    hp = Hp_g + blk * (HEADDIM * DSTATE);
    float g = expf(cs_g[blk * QLEN + (QLEN - 1)]);
#pragma unroll
    for (int i = 0; i < 16; ++i) {
      int e = i * 256 + t;
      hp[e] = f32_to_bf16(S[i]);      // state at chunk start
      S[i]  = S[i] * g + hc[e];       // advance to chunk end
    }
  }
}

// ---------------- chunked scan pass C: per-chunk outputs via 3 WMMAs ---------------------
// Y[t][p] = sum_{s<=t} exp(cs[t]-cs[s])*dt[s]*(C_t.B_s)*x[s][p]
//         + exp(cs[t]) * sum_n C_t[n]*h0[p][n]  +  D*x[t][p]
__global__ __launch_bounds__(256)
void chunk_output_kernel(const float* __restrict__ xconv,
                         const float* __restrict__ dtv,
                         const float* __restrict__ cs_g,
                         const uint16_t* __restrict__ Hp_g,
                         const float* __restrict__ Dp,
                         float* __restrict__ y) {
  const int blk = blockIdx.x;
  const int c   = blk % CHUNKS;
  const int bh  = blk / CHUNKS;
  const int hh  = bh % NHEADS;
  const int b   = bh / NHEADS;
  const int t   = threadIdx.x;
  const int lane = t & 31, wave = t >> 5;
  const int lm = lane & 15, kg = lane >> 4;
  const int mt = wave & 3, nt0 = (wave >> 2) * 2;

  __shared__ float csL[QLEN];
  __shared__ float dtL[QLEN];
  __shared__ __attribute__((aligned(16))) uint16_t Ct[QLEN * LDS2];  // [t][n]
  __shared__ __attribute__((aligned(16))) uint16_t Bt[QLEN * LDS2];  // [s][n]
  __shared__ __attribute__((aligned(16))) uint16_t xT[QLEN * LDS2];  // [p][s]
  __shared__ __attribute__((aligned(16))) uint16_t Gm[QLEN * LDS2];  // [t][s]
  __shared__ __attribute__((aligned(16))) uint16_t Hp[QLEN * LDS2];  // [p][n]

  const size_t row0 = (size_t)b * SEQ + c * QLEN;
  if (t < QLEN) {
    csL[t] = cs_g[(size_t)blk * QLEN + t];
    dtL[t] = dtv[(row0 + t) * NHEADS + hh];
  }
  const uint16_t* hp_g = Hp_g + (size_t)blk * (HEADDIM * DSTATE);
  for (int e = t; e < QLEN * QLEN; e += 256) {
    int r = e >> 6, d = e & 63;
    Ct[r * LDS2 + d] = f32_to_bf16(xconv[(row0 + r) * CONVDIM + DINNER + DSTATE + d]);
    Bt[r * LDS2 + d] = f32_to_bf16(xconv[(row0 + r) * CONVDIM + DINNER + d]);
    xT[d * LDS2 + r] = f32_to_bf16(xconv[(row0 + r) * CONVDIM + hh * HEADDIM + d]);
    Hp[r * LDS2 + d] = hp_g[e];
  }
  __syncthreads();

  const v8f vzero = {0.f,0.f,0.f,0.f,0.f,0.f,0.f,0.f};

  // S = C @ B^T, then mask/scale into G (bf16 in LDS)
  v8f accS[2] = {vzero, vzero};
  wave_gemm64(Ct, Bt, mt, nt0, accS, lane);
#pragma unroll
  for (int j = 0; j < 2; ++j) {
    int s = (nt0 + j) * 16 + lm;
#pragma unroll
    for (int v = 0; v < 8; ++v) {
      int tt = mt * 16 + v + 8 * kg;
      float gv = 0.f;
      if (s <= tt) gv = accS[j][v] * expf(csL[tt] - csL[s]) * dtL[s];
      Gm[tt * LDS2 + s] = f32_to_bf16(gv);
    }
  }
  __syncthreads();

  v8f accY[2] = {vzero, vzero};
  wave_gemm64(Gm, xT, mt, nt0, accY, lane);   // Y_intra[t][p]
  v8f accI[2] = {vzero, vzero};
  wave_gemm64(Ct, Hp, mt, nt0, accI, lane);   // Y_inter[t][p] (unscaled)

  const float Dh = Dp[hh];
#pragma unroll
  for (int j = 0; j < 2; ++j) {
    int p = (nt0 + j) * 16 + lm;
#pragma unroll
    for (int v = 0; v < 8; ++v) {
      int tt = mt * 16 + v + 8 * kg;
      float xval = xconv[(row0 + tt) * CONVDIM + hh * HEADDIM + p];
      float yv = accY[j][v] + expf(csL[tt]) * accI[j][v] + Dh * xval;
      y[(row0 + tt) * DINNER + hh * HEADDIM + p] = yv;
    }
  }
}

// ---------------- gating (y * silu(z)) + RMSNorm + emit bf16 ----------------
__global__ __launch_bounds__(256)
void gate_norm_kernel(const float* __restrict__ zx,
                      const float* __restrict__ y,
                      const float* __restrict__ norm_w,
                      uint16_t* __restrict__ ybf) {
  const int r = blockIdx.x;
  const int t = threadIdx.x;
  __shared__ float red[256];
  float g[8];
  float ss = 0.f;
#pragma unroll
  for (int jj = 0; jj < 8; ++jj) {
    int i = jj * 256 + t;
    float z  = zx[(size_t)r * NPAD + i];
    float yv = y[(size_t)r * DINNER + i];
    float gv = yv * (z * sigmoidf_(z));
    g[jj] = gv;
    ss += gv * gv;
  }
  red[t] = ss;
  __syncthreads();
  for (int s = 128; s > 0; s >>= 1) {
    if (t < s) red[t] += red[t + s];
    __syncthreads();
  }
  float scale = rsqrtf(red[0] / (float)DINNER + EPSV);
#pragma unroll
  for (int jj = 0; jj < 8; ++jj) {
    int i = jj * 256 + t;
    ybf[(size_t)r * DINNER + i] = f32_to_bf16(g[jj] * scale * norm_w[i]);
  }
}

// ---------------- host launch ----------------
extern "C" void kernel_launch(void* const* d_in, const int* in_sizes, int n_in,
                              void* d_out, int out_size, void* d_ws, size_t ws_size,
                              hipStream_t stream) {
  const float* X       = (const float*)d_in[0];
  const float* W_in    = (const float*)d_in[1];
  const float* conv_w  = (const float*)d_in[2];
  const float* conv_b  = (const float*)d_in[3];
  const float* dt_bias = (const float*)d_in[4];
  const float* A_log   = (const float*)d_in[5];
  const float* Dp      = (const float*)d_in[6];
  const float* norm_w  = (const float*)d_in[7];
  const float* W_out   = (const float*)d_in[8];
  float* out = (float*)d_out;

  char* ws = (char*)d_ws;
  size_t off = 0;
  auto carve = [&](size_t bytes) -> void* {
    void* p = ws + off;
    off += (bytes + 255) & ~(size_t)255;
    return p;
  };
  uint16_t* Xbf    = (uint16_t*)carve((size_t)MTOT * DMODEL * 2);
  uint16_t* Winbf  = (uint16_t*)carve((size_t)NPAD * DMODEL * 2);
  uint16_t* Woutbf = (uint16_t*)carve((size_t)DMODEL * DINNER * 2);
  float*    zx     = (float*)carve((size_t)MTOT * NPAD * 4);
  float*    xconv  = (float*)carve((size_t)MTOT * CONVDIM * 4);
  float*    dtv    = (float*)carve((size_t)MTOT * NHEADS * 4);
  float*    ldAv   = (float*)carve((size_t)MTOT * NHEADS * 4);
  float*    cs_g   = (float*)carve((size_t)NCHUNK * QLEN * 4);
  float*    Hc_g   = (float*)carve((size_t)NCHUNK * HEADDIM * DSTATE * 4);
  uint16_t* Hp_g   = (uint16_t*)carve((size_t)NCHUNK * HEADDIM * DSTATE * 2);
  float*    yb     = (float*)carve((size_t)MTOT * DINNER * 4);
  uint16_t* ybf    = (uint16_t*)carve((size_t)MTOT * DINNER * 2);

  // 1) conversions
  {
    int n = MTOT * DMODEL;
    cvt_bf16_kernel<<<(n + 255) / 256, 256, 0, stream>>>(X, Xbf, n);
    int nt = NPAD * DMODEL, nv = DINPROJ * DMODEL;
    cvt_bf16_pad_kernel<<<(nt + 255) / 256, 256, 0, stream>>>(W_in, Winbf, nv, nt);
    int nw = DMODEL * DINNER;
    cvt_bf16_kernel<<<(nw + 255) / 256, 256, 0, stream>>>(W_out, Woutbf, nw);
  }

  // 2) GEMM1: zx[M, NPAD] = X @ W_in_pad^T
  {
    dim3 grid(MTOT / BM, NPAD / BN);
    gemm_bf16_wmma_kernel<<<grid, 256, 0, stream>>>(Xbf, Winbf, zx, nullptr,
                                                    NPAD, DMODEL);
  }

  // 3) conv + silu ; dt/ldA precompute
  {
    int n = MTOT * CONVDIM;
    conv_silu_kernel<<<(n + 255) / 256, 256, 0, stream>>>(zx, conv_w, conv_b, xconv);
    int nd = MTOT * NHEADS;
    dt_kernel<<<(nd + 255) / 256, 256, 0, stream>>>(zx, dt_bias, A_log, dtv, ldAv);
  }

  // 4) chunked SSM scan (SSD): A (parallel WMMA) -> B (short recurrence) -> C (parallel WMMA)
  chunk_state_kernel<<<NCHUNK, 256, 0, stream>>>(xconv, dtv, ldAv, cs_g, Hc_g);
  chunk_recur_kernel<<<BATCH * NHEADS, 256, 0, stream>>>(Hc_g, cs_g, Hp_g);
  chunk_output_kernel<<<NCHUNK, 256, 0, stream>>>(xconv, dtv, cs_g, Hp_g, Dp, yb);

  // 5) gating + RMSNorm -> bf16
  gate_norm_kernel<<<MTOT, 256, 0, stream>>>(zx, yb, norm_w, ybf);

  // 6) GEMM2 with fused residual: out = ybf @ W_out^T + X
  {
    dim3 grid(MTOT / BM, DMODEL / BN);
    gemm_bf16_wmma_kernel<<<grid, 256, 0, stream>>>(ybf, Woutbf, out, X,
                                                    DMODEL, DINNER);
  }
}
